// MultiAttentionHead_62852551409722
// MI455X (gfx1250) — matrix operand order repaired and checked by reading, not compiled, not run on hardware
//
#include <hip/hip_runtime.h>

#define BATCH 4
#define SEQ   2048
#define DIM   1024
#define NH    16
#define HDIM  64

typedef __attribute__((ext_vector_type(8)))  __bf16 v8bf;
typedef __attribute__((ext_vector_type(16))) __bf16 v16bf;
typedef __attribute__((ext_vector_type(8)))  float  v8f;

static __device__ __forceinline__ __bf16 f2bf(float f) {
  unsigned u = __builtin_bit_cast(unsigned, f);
  u += 0x7FFFu + ((u >> 16) & 1u);           // round to nearest even
  unsigned short h = (unsigned short)(u >> 16);
  return __builtin_bit_cast(__bf16, h);
}

static __device__ __forceinline__ v16bf cat8(v8bf a, v8bf b) {
  v16bf r;
#pragma unroll
  for (int i = 0; i < 8; ++i) { r[i] = a[i]; r[i + 8] = b[i]; }
  return r;
}

static __device__ __forceinline__ v8f wmma_bf16(v16bf a, v16bf b, v8f c) {
  return __builtin_amdgcn_wmma_f32_16x16x32_bf16(
      /*neg_a=*/false, a, /*neg_b=*/false, b,
      /*c_mod=*/(short)0, c, /*reuse_a=*/false, /*reuse_b=*/false);
}

// ---------------------------------------------------------------------------
// Prepack: x -> bf16
__global__ __launch_bounds__(256) void k_cvt_x(const float* __restrict__ x,
                                               __bf16* __restrict__ xb, int n) {
  int i = blockIdx.x * blockDim.x + threadIdx.x;
  if (i < n) xb[i] = f2bf(x[i]);
}

// Prepack weights: Wq/Wk/Wv [H,D,HD] -> WT_all [3][H][HD][D] bf16 (transposed
// so B-fragments read contiguous K); Wp [D,D] -> Wpb [n][k] bf16 (unchanged).
__global__ __launch_bounds__(256) void k_prep_w(
    const float* __restrict__ Wq, const float* __restrict__ Wk,
    const float* __restrict__ Wv, const float* __restrict__ Wp,
    __bf16* __restrict__ WT_all, __bf16* __restrict__ Wpb) {
  int i = blockIdx.x * blockDim.x + threadIdx.x;  // < 4 * 1M
  int which = i >> 20;
  int rem = i & 0xFFFFF;
  if (which < 3) {
    const float* W = (which == 0) ? Wq : (which == 1) ? Wk : Wv;
    int h = rem >> 16;            // HDIM*DIM = 65536
    int r2 = rem & 0xFFFF;
    int n = r2 >> 10;             // 0..63
    int k = r2 & 1023;            // 0..1023
    WT_all[(size_t)which * (NH * HDIM * DIM) + (size_t)h * (HDIM * DIM) +
           (size_t)n * DIM + k] = f2bf(W[(size_t)h * (DIM * HDIM) + (size_t)k * HDIM + n]);
  } else {
    Wpb[rem] = f2bf(Wp[rem]);
  }
}

// ---------------------------------------------------------------------------
// QKV projection: per-wave 32x64 tile (2 M-subtiles x 4 N-subtiles = 8 wmma
// per K-step sharing fragments). q,k stored [B,H,T,HD] bf16; v stored
// transposed [B,H,HD,T] bf16.
__global__ __launch_bounds__(256) void k_qkv(
    const __bf16* __restrict__ xb, const __bf16* __restrict__ WT_all,
    const float* __restrict__ bq, const float* __restrict__ bk,
    const float* __restrict__ bv, __bf16* __restrict__ Qb,
    __bf16* __restrict__ Kb, __bf16* __restrict__ VTr) {
  int lane = threadIdx.x & 31;
  int gw = (int)((blockIdx.x * blockDim.x + threadIdx.x) >> 5);
  int mt  = gw & 255;          // (B*T)/32 row tiles
  int h   = (gw >> 8) & 15;
  int mat = gw >> 12;          // 0=q 1=k 2=v
  int lh = lane >> 4, ll = lane & 15;

  const __bf16* WT = WT_all + (size_t)mat * (NH * HDIM * DIM) + (size_t)h * (HDIM * DIM);
  const __bf16* arow0 = xb + (size_t)(mt * 32 + ll) * DIM;
  const __bf16* arow1 = xb + (size_t)(mt * 32 + 16 + ll) * DIM;
  const __bf16* brow0 = WT + (size_t)(0 * 16 + ll) * DIM;
  const __bf16* brow1 = WT + (size_t)(1 * 16 + ll) * DIM;
  const __bf16* brow2 = WT + (size_t)(2 * 16 + ll) * DIM;
  const __bf16* brow3 = WT + (size_t)(3 * 16 + ll) * DIM;

  v8f acc[2][4] = {};
  for (int k0 = 0; k0 < DIM; k0 += 32) {
    int ca = k0 + lh * 16;
    int cb = k0 + lh * 16;
    v16bf a0 = cat8(*(const v8bf*)(arow0 + ca), *(const v8bf*)(arow0 + ca + 8));
    v16bf a1 = cat8(*(const v8bf*)(arow1 + ca), *(const v8bf*)(arow1 + ca + 8));
    v16bf b0 = *(const v16bf*)(brow0 + cb);
    v16bf b1 = *(const v16bf*)(brow1 + cb);
    v16bf b2 = *(const v16bf*)(brow2 + cb);
    v16bf b3 = *(const v16bf*)(brow3 + cb);
    acc[0][0] = wmma_bf16(a0, b0, acc[0][0]);
    acc[0][1] = wmma_bf16(a0, b1, acc[0][1]);
    acc[0][2] = wmma_bf16(a0, b2, acc[0][2]);
    acc[0][3] = wmma_bf16(a0, b3, acc[0][3]);
    acc[1][0] = wmma_bf16(a1, b0, acc[1][0]);
    acc[1][1] = wmma_bf16(a1, b1, acc[1][1]);
    acc[1][2] = wmma_bf16(a1, b2, acc[1][2]);
    acc[1][3] = wmma_bf16(a1, b3, acc[1][3]);
  }

  const float* bias = (mat == 0) ? bq : (mat == 1) ? bk : bv;
#pragma unroll
  for (int nt = 0; nt < 4; ++nt) {
    int n = nt * 16 + ll;
    float bn = bias[h * HDIM + n];
#pragma unroll
    for (int ms = 0; ms < 2; ++ms) {
#pragma unroll
      for (int r = 0; r < 8; ++r) {
        int grow = mt * 32 + ms * 16 + lh * 8 + r;
        int b = grow >> 11;        // /SEQ
        int t = grow & (SEQ - 1);
        __bf16 val = f2bf(acc[ms][nt][r] + bn);
        size_t bh = (size_t)(b * NH + h);
        if (mat == 0)      Qb[(bh * SEQ + t) * HDIM + n] = val;
        else if (mat == 1) Kb[(bh * SEQ + t) * HDIM + n] = val;
        else               VTr[(bh * HDIM + n) * SEQ + t] = val;
      }
    }
  }
}

// ---------------------------------------------------------------------------
// Flash attention: one wave per (b, h, 16-row query tile); 32-key chunks.
__global__ __launch_bounds__(256) void k_attn(
    const __bf16* __restrict__ Qb, const __bf16* __restrict__ Kb,
    const __bf16* __restrict__ VTr, __bf16* __restrict__ xo) {
  __shared__ __bf16 psh[8][16 * 32];   // per-wave P staging (C-layout -> A-frag)
  int lane = threadIdx.x & 31;
  int wv = threadIdx.x >> 5;
  int gw = blockIdx.x * 8 + wv;
  int qt = gw & 127;            // SEQ/16
  int h  = (gw >> 7) & 15;
  int b  = gw >> 11;
  int lh = lane >> 4, ll = lane & 15;
  size_t bh = (size_t)(b * NH + h);

  const __bf16* qrow = Qb + (bh * SEQ + qt * 16 + ll) * HDIM;
  v16bf aq0 = cat8(*(const v8bf*)(qrow + lh * 16),      *(const v8bf*)(qrow + lh * 16 + 8));
  v16bf aq1 = cat8(*(const v8bf*)(qrow + 32 + lh * 16), *(const v8bf*)(qrow + 32 + lh * 16 + 8));

  v8f o0 = {}, o1 = {}, o2 = {}, o3 = {};
  float mrow[8], lrow[8];
#pragma unroll
  for (int r = 0; r < 8; ++r) { mrow[r] = -3.0e38f; lrow[r] = 0.0f; }
  const float scale = 0.125f;   // HD^-0.5

  int nch = (qt >> 1) + 1;      // 32-key chunks covering keys 0..qt*16+15
  for (int c0 = 0; c0 < nch; ++c0) {
    int kt0 = c0 * 32;
    float s0a[8], s1a[8], p0[8], p1[8];
    {
      const __bf16* kr0 = Kb + (bh * SEQ + (kt0 + ll)) * HDIM;
      v8f s = {};
      s = wmma_bf16(aq0, *(const v16bf*)(kr0 + lh * 16), s);
      s = wmma_bf16(aq1, *(const v16bf*)(kr0 + 32 + lh * 16), s);
      const __bf16* kr1 = Kb + (bh * SEQ + (kt0 + 16 + ll)) * HDIM;
      v8f s2 = {};
      s2 = wmma_bf16(aq0, *(const v16bf*)(kr1 + lh * 16), s2);
      s2 = wmma_bf16(aq1, *(const v16bf*)(kr1 + 32 + lh * 16), s2);
      int key0 = kt0 + ll, key1 = kt0 + 16 + ll;
#pragma unroll
      for (int r = 0; r < 8; ++r) {
        int row = qt * 16 + lh * 8 + r;
        s0a[r] = (key0 <= row) ? s[r] * scale : -3.0e38f;
        s1a[r] = (key1 <= row) ? s2[r] * scale : -3.0e38f;
      }
    }
#pragma unroll
    for (int r = 0; r < 8; ++r) {
      float mx = fmaxf(s0a[r], s1a[r]);
      mx = fmaxf(mx, __shfl_xor(mx, 1, 32));
      mx = fmaxf(mx, __shfl_xor(mx, 2, 32));
      mx = fmaxf(mx, __shfl_xor(mx, 4, 32));
      mx = fmaxf(mx, __shfl_xor(mx, 8, 32));
      float mnew = fmaxf(mrow[r], mx);
      float alpha = __expf(mrow[r] - mnew);
      mrow[r] = mnew;
      float a0 = __expf(s0a[r] - mnew);
      float a1 = __expf(s1a[r] - mnew);
      p0[r] = a0; p1[r] = a1;
      float rs = a0 + a1;
      rs += __shfl_xor(rs, 1, 32);
      rs += __shfl_xor(rs, 2, 32);
      rs += __shfl_xor(rs, 4, 32);
      rs += __shfl_xor(rs, 8, 32);
      lrow[r] = lrow[r] * alpha + rs;
      o0[r] *= alpha; o1[r] *= alpha; o2[r] *= alpha; o3[r] *= alpha;
    }
    // C-layout -> A-fragment via per-wave LDS (same-wave DS ops are in order)
    __bf16* ps = &psh[wv][0];
#pragma unroll
    for (int r = 0; r < 8; ++r) {
      int row = lh * 8 + r;
      ps[row * 32 + ll]      = f2bf(p0[r]);
      ps[row * 32 + 16 + ll] = f2bf(p1[r]);
    }
    v16bf ap = cat8(*(const v8bf*)(ps + ll * 32 + lh * 16),
                    *(const v8bf*)(ps + ll * 32 + lh * 16 + 8));
    const __bf16* vb = VTr + (bh * HDIM) * SEQ + kt0 + lh * 16;
    o0 = wmma_bf16(ap, *(const v16bf*)(vb + (size_t)(0 * 16 + ll) * SEQ), o0);
    o1 = wmma_bf16(ap, *(const v16bf*)(vb + (size_t)(1 * 16 + ll) * SEQ), o1);
    o2 = wmma_bf16(ap, *(const v16bf*)(vb + (size_t)(2 * 16 + ll) * SEQ), o2);
    o3 = wmma_bf16(ap, *(const v16bf*)(vb + (size_t)(3 * 16 + ll) * SEQ), o3);
  }
#pragma unroll
  for (int r = 0; r < 8; ++r) {
    float inv = 1.0f / lrow[r];
    int grow = qt * 16 + lh * 8 + r;
    __bf16* orow = xo + ((size_t)b * SEQ + grow) * DIM + h * HDIM;
    orow[ll]      = f2bf(o0[r] * inv);
    orow[16 + ll] = f2bf(o1[r] * inv);
    orow[32 + ll] = f2bf(o2[r] * inv);
    orow[48 + ll] = f2bf(o3[r] * inv);
  }
}

// ---------------------------------------------------------------------------
// Output projection: y = xo @ Wp^T + bp (f32 out). Per-wave 32x64 tile.
__global__ __launch_bounds__(256) void k_proj(
    const __bf16* __restrict__ xo, const __bf16* __restrict__ Wpb,
    const float* __restrict__ bp, float* __restrict__ out) {
  int lane = threadIdx.x & 31;
  int gw = (int)((blockIdx.x * blockDim.x + threadIdx.x) >> 5);
  int mt  = gw & 255;           // (B*T)/32 row tiles
  int nt4 = gw >> 8;            // 0..15 (groups of 64 cols)
  int lh = lane >> 4, ll = lane & 15;
  const __bf16* arow0 = xo + (size_t)(mt * 32 + ll) * DIM;
  const __bf16* arow1 = xo + (size_t)(mt * 32 + 16 + ll) * DIM;
  const __bf16* brow0 = Wpb + (size_t)(nt4 * 64 + 0 * 16 + ll) * DIM;
  const __bf16* brow1 = Wpb + (size_t)(nt4 * 64 + 1 * 16 + ll) * DIM;
  const __bf16* brow2 = Wpb + (size_t)(nt4 * 64 + 2 * 16 + ll) * DIM;
  const __bf16* brow3 = Wpb + (size_t)(nt4 * 64 + 3 * 16 + ll) * DIM;

  v8f acc[2][4] = {};
  for (int k0 = 0; k0 < DIM; k0 += 32) {
    int ca = k0 + lh * 16;
    int cb = k0 + lh * 16;
    v16bf a0 = cat8(*(const v8bf*)(arow0 + ca), *(const v8bf*)(arow0 + ca + 8));
    v16bf a1 = cat8(*(const v8bf*)(arow1 + ca), *(const v8bf*)(arow1 + ca + 8));
    v16bf b0 = *(const v16bf*)(brow0 + cb);
    v16bf b1 = *(const v16bf*)(brow1 + cb);
    v16bf b2 = *(const v16bf*)(brow2 + cb);
    v16bf b3 = *(const v16bf*)(brow3 + cb);
    acc[0][0] = wmma_bf16(a0, b0, acc[0][0]);
    acc[0][1] = wmma_bf16(a0, b1, acc[0][1]);
    acc[0][2] = wmma_bf16(a0, b2, acc[0][2]);
    acc[0][3] = wmma_bf16(a0, b3, acc[0][3]);
    acc[1][0] = wmma_bf16(a1, b0, acc[1][0]);
    acc[1][1] = wmma_bf16(a1, b1, acc[1][1]);
    acc[1][2] = wmma_bf16(a1, b2, acc[1][2]);
    acc[1][3] = wmma_bf16(a1, b3, acc[1][3]);
  }
#pragma unroll
  for (int nt = 0; nt < 4; ++nt) {
    int n = nt4 * 64 + nt * 16 + ll;
    float bn = bp[n];
#pragma unroll
    for (int ms = 0; ms < 2; ++ms) {
#pragma unroll
      for (int r = 0; r < 8; ++r) {
        int grow = mt * 32 + ms * 16 + lh * 8 + r;
        out[(size_t)grow * DIM + n] = acc[ms][nt][r] + bn;
      }
    }
  }
}

// ---------------------------------------------------------------------------
extern "C" void kernel_launch(void* const* d_in, const int* in_sizes, int n_in,
                              void* d_out, int out_size, void* d_ws, size_t ws_size,
                              hipStream_t stream) {
  const float* x  = (const float*)d_in[0];
  const float* Wq = (const float*)d_in[1];
  const float* bq = (const float*)d_in[2];
  const float* Wk = (const float*)d_in[3];
  const float* bk = (const float*)d_in[4];
  const float* Wv = (const float*)d_in[5];
  const float* bv = (const float*)d_in[6];
  const float* Wp = (const float*)d_in[7];
  const float* bp = (const float*)d_in[8];
  float* out = (float*)d_out;

  char* ws = (char*)d_ws;
  size_t off = 0;
  __bf16* xb     = (__bf16*)(ws + off); off += (size_t)BATCH * SEQ * DIM * 2;        // 16 MB
  __bf16* WT_all = (__bf16*)(ws + off); off += (size_t)3 * NH * HDIM * DIM * 2;      // 6 MB
  __bf16* Wpb    = (__bf16*)(ws + off); off += (size_t)DIM * DIM * 2;                // 2 MB
  __bf16* Qb     = (__bf16*)(ws + off); off += (size_t)BATCH * NH * SEQ * HDIM * 2;  // 16 MB
  __bf16* Kb     = (__bf16*)(ws + off); off += (size_t)BATCH * NH * SEQ * HDIM * 2;  // 16 MB
  __bf16* VTr    = (__bf16*)(ws + off); off += (size_t)BATCH * NH * HDIM * SEQ * 2;  // 16 MB
  __bf16* xo     = (__bf16*)(ws + off); off += (size_t)BATCH * SEQ * DIM * 2;        // 16 MB

  const int nx = BATCH * SEQ * DIM;                       // 8,388,608
  k_cvt_x<<<(nx + 255) / 256, 256, 0, stream>>>(x, xb, nx);
  k_prep_w<<<(4 * 1048576) / 256, 256, 0, stream>>>(Wq, Wk, Wv, Wp, WT_all, Wpb);

  // 3 mats * 16 heads * 256 mtiles waves, 8 waves/block
  k_qkv<<<1536, 256, 0, stream>>>(xb, WT_all, bq, bk, bv, Qb, Kb, VTr);

  // B*H*(T/16) = 8192 waves, 8 waves/block
  k_attn<<<1024, 256, 0, stream>>>(Qb, Kb, VTr, xo);

  // 256 mtiles * 16 ntile-groups = 4096 waves, 8 waves/block
  k_proj<<<512, 256, 0, stream>>>(xo, Wpb, bp, out);
}
